// Model_68539088110265
// MI455X (gfx1250) — matrix operand order repaired and checked by reading, not compiled, not run on hardware
//
#include <hip/hip_runtime.h>
#include <hip/hip_bf16.h>

// ---------------------------------------------------------------------------
// Model dims: B=8, T=512, N=64, D=128, PRED=96, HEADS=8, dh=16, HID=256
// Rank-collapsed pipeline (NUM_BLOCKS=NUM_LAYERS=1):
//   decomp -> rfft64 -> CEM (rank-1, scalar attention) -> G(b,n,h,t)
//   -> rDFT512 as WMMA-f16 GEMM -> TEM (rank-8 attention) -> Z
//   -> irDFT512 as WMMA-f16 GEMM -> output GEMM; trend MLP via WMMA-bf16.
// GEMM tiles staged via the Tensor Data Mover (tensor_load_to_lds +
// s_wait_tensorcnt) when the builtin is available; fragments are packed in
// the exact CDNA5 per-lane WMMA layout and fetched with ds_load_b128.
// ---------------------------------------------------------------------------

typedef __bf16 bf16;
typedef __attribute__((ext_vector_type(16))) __bf16    v16bf;
typedef __attribute__((ext_vector_type(16))) _Float16  v16h;
typedef __attribute__((ext_vector_type(8)))  float     v8f;
typedef __attribute__((ext_vector_type(4)))  unsigned  v4u;

__device__ __forceinline__ bf16 f2bf(float f) { return (bf16)f; }

__device__ __forceinline__ v8f wmma_any(v16bf a, v16bf b, v8f c) {
  return __builtin_amdgcn_wmma_f32_16x16x32_bf16(false, a, false, b, (short)0, c,
                                                 false, false);
}
__device__ __forceinline__ v8f wmma_any(v16h a, v16h b, v8f c) {
  return __builtin_amdgcn_wmma_f32_16x16x32_f16(false, a, false, b, (short)0, c,
                                                false, false);
}

template <typename T> union Frag;
template <> union Frag<__bf16>   { v16bf v; v4u u[2]; };
template <> union Frag<_Float16> { v16h  v; v4u u[2]; };

template <typename T>
__device__ __forceinline__ unsigned pack2(float a, float b) {
  union { T h; unsigned short s; } ua, ub;
  ua.h = (T)a; ub.h = (T)b;
  return (unsigned)ua.s | ((unsigned)ub.s << 16);
}

// ---------------------------------------------------------------------------
// Tensor Data Mover helpers (guarded: builtin arity differs per toolchain)
// ---------------------------------------------------------------------------
#define HAVE_TDM __has_builtin(__builtin_amdgcn_tensor_load_to_lds)

#if HAVE_TDM
typedef __attribute__((ext_vector_type(4))) unsigned tdm_v4u;
typedef __attribute__((ext_vector_type(8))) int      tdm_v8i;
typedef __attribute__((ext_vector_type(4))) int      tdm_v4i;

__device__ __forceinline__ unsigned lds_offset_of(const void* p) {
  return (unsigned)(unsigned long long)(__attribute__((address_space(3))) const char*)p;
}

// 2D f32 tile load: tile_w x tile_h from row-major tensor (pitch row_stride
// elements), zero-fill past (tens_w, tens_h) measured from the tile origin.
__device__ __forceinline__ void tdm_load_2d(const void* gaddr, unsigned lds_off,
                                            unsigned tile_w, unsigned tile_h,
                                            unsigned tens_w, unsigned tens_h,
                                            unsigned row_stride) {
  unsigned long long ga = (unsigned long long)gaddr;
  tdm_v4u g0;
  g0[0] = 1u;                                                  // count=1 (user)
  g0[1] = lds_off;                                             // lds_addr
  g0[2] = (unsigned)(ga & 0xFFFFFFFFu);                        // global_addr lo
  g0[3] = (unsigned)((ga >> 32) & 0x01FFFFFFu) | (2u << 30);   // addr hi | type=2
  tdm_v8i g1;
  g1[0] = (int)(2u << 16);                                     // data_size=4B
  g1[1] = (int)((tens_w & 0xFFFFu) << 16);                     // tensor_dim0 lo
  g1[2] = (int)((tens_w >> 16) | ((tens_h & 0xFFFFu) << 16));  // dim0 hi|dim1 lo
  g1[3] = (int)((tens_h >> 16) | (tile_w << 16));              // dim1 hi|tile0
  g1[4] = (int)(tile_h & 0xFFFFu);                             // tile1 (tile2=0)
  g1[5] = (int)row_stride;                                     // dim0_stride lo
  g1[6] = 0;
  g1[7] = 0;
  tdm_v4i zz = {0, 0, 0, 0};
#if __clang_major__ >= 23
  tdm_v8i z8 = {0, 0, 0, 0, 0, 0, 0, 0};
  __builtin_amdgcn_tensor_load_to_lds(g0, g1, zz, zz, z8, 0);
#else
  __builtin_amdgcn_tensor_load_to_lds(g0, g1, zz, zz, 0);
#endif
}
#endif  // HAVE_TDM

// ---------------------------------------------------------------------------
// Kernel 1: series decomposition (two moving averages + softmax mixing)
// ---------------------------------------------------------------------------
__global__ void decomp_kernel(const float* __restrict__ x,
                              const float* __restrict__ dec_w,
                              const float* __restrict__ dec_b,
                              float* __restrict__ res,
                              float* __restrict__ trendT) {
  const int b = blockIdx.x >> 6;  // N=64
  const int n = blockIdx.x & 63;
  __shared__ float xs[512];
  for (int t = threadIdx.x; t < 512; t += blockDim.x)
    xs[t] = x[(b * 512 + t) * 64 + n];
  __syncthreads();
  const float w0 = dec_w[0], w1 = dec_w[1];
  const float b0 = dec_b[0], b1 = dec_b[1];
  for (int t = threadIdx.x; t < 512; t += blockDim.x) {
    float s0 = 0.f, s1 = 0.f;
    for (int j = -24; j <= 24; ++j) {
      int idx = t + j;
      idx = idx < 0 ? 0 : (idx > 511 ? 511 : idx);  // edge-replicate pad
      float v = xs[idx];
      s1 += v;
      if (j >= -8 && j <= 8) s0 += v;
    }
    float m0 = s0 * (1.f / 17.f), m1 = s1 * (1.f / 49.f);
    float xv = xs[t];
    float e0 = __expf(xv * w0 + b0), e1 = __expf(xv * w1 + b1);
    float tr = (m0 * e0 + m1 * e1) / (e0 + e1);
    res[(b * 512 + t) * 64 + n] = xv - tr;
    trendT[(b * 64 + n) * 512 + t] = tr;
  }
}

// ---------------------------------------------------------------------------
// Kernel 2: 64-pt rDFT of res rows via LDS twiddle LUT -> rf (complex, k<32)
// ---------------------------------------------------------------------------
__global__ void rfft64_kernel(const float* __restrict__ res,
                              float* __restrict__ rf_re,
                              float* __restrict__ rf_im) {
  __shared__ float xr[8][64];
  __shared__ float twc[64], tws[64];
  const int tid = threadIdx.x;  // 256 = 8 rows x 32 k
  if (tid < 64) {
    float th = -(6.28318530717958647f / 64.f) * (float)tid;
    __sincosf(th, &tws[tid], &twc[tid]);
  }
  for (int i = tid; i < 512; i += 256) {
    int r = i >> 6, n = i & 63;
    xr[r][n] = res[(blockIdx.x * 8 + r) * 64 + n];
  }
  __syncthreads();
  const int lr = tid >> 5;
  const int k = tid & 31;
  const int row = blockIdx.x * 8 + lr;
  float sr = 0.f, si = 0.f;
  for (int n = 0; n < 64; ++n) {
    int idx = (k * n) & 63;
    float v = xr[lr][n];
    sr += v * twc[idx];
    si += v * tws[idx];
  }
  rf_re[row * 32 + k] = sr;
  rf_im[row * 32 + k] = si;
}

// ---------------------------------------------------------------------------
// Kernel 3: rank-collapsed constants (1 block, 128 threads)
// ---------------------------------------------------------------------------
__global__ void prep_kernel(const float* __restrict__ emb,
                            const float* __restrict__ cWq, const float* __restrict__ cWk,
                            const float* __restrict__ cWv, const float* __restrict__ cWo,
                            const float* __restrict__ tWq, const float* __restrict__ tWk,
                            const float* __restrict__ tWv, const float* __restrict__ tWo,
                            const float* __restrict__ dr_w,
                            float* __restrict__ c_out, float* __restrict__ Pq,
                            float* __restrict__ Pk, float* __restrict__ Dv) {
  __shared__ float qe[128], ke[128], ve[128];
  __shared__ float wot[8][128];
  __shared__ float pvs[8][128];
  __shared__ float wod[128];
  const int e = threadIdx.x;
  float aq = 0.f, ak = 0.f, av = 0.f;
  for (int d = 0; d < 128; ++d) {
    float em = emb[d];
    aq += em * cWq[d * 128 + e];
    ak += em * cWk[d * 128 + e];
    av += em * cWv[d * 128 + e];
  }
  qe[e] = aq; ke[e] = ak; ve[e] = av;
  __syncthreads();
  if (e < 8) {
    float c = 0.f;
    for (int d = 0; d < 16; ++d) c += qe[e * 16 + d] * ke[e * 16 + d];
    c_out[e] = c;
  }
  for (int h = 0; h < 8; ++h) {
    float s = 0.f;
    for (int d = 0; d < 16; ++d) s += ve[h * 16 + d] * cWo[(h * 16 + d) * 128 + e];
    wot[h][e] = s;
  }
  {
    float s = 0.f;
    for (int d = 0; d < 128; ++d) s += tWo[e * 128 + d] * dr_w[d];
    wod[e] = s;
  }
  __syncthreads();
  for (int h = 0; h < 8; ++h) {
    float sq = 0.f, sk = 0.f, sv = 0.f;
    for (int ep = 0; ep < 128; ++ep) {
      float w = wot[h][ep];
      sq += w * tWq[ep * 128 + e];
      sk += w * tWk[ep * 128 + e];
      sv += w * tWv[ep * 128 + e];
    }
    Pq[h * 128 + e] = sq;
    Pk[h * 128 + e] = sk;
    pvs[h][e] = sv;
  }
  __syncthreads();
  if (e < 64) {
    int h = e >> 3, h1 = e & 7;
    float s = 0.f;
    for (int d = 0; d < 16; ++d) s += pvs[h1][h * 16 + d] * wod[h * 16 + d];
    Dv[h * 8 + h1] = s;
  }
}

// ---------------------------------------------------------------------------
// Kernel 4: CEM attention (rank-1 collapsed) + 64-pt irDFT (LDS twiddle LUT)
// ---------------------------------------------------------------------------
__global__ void cem_attn_kernel(const float* __restrict__ rf_re,
                                const float* __restrict__ rf_im,
                                const float* __restrict__ c_h,
                                float* __restrict__ G) {
  const int bt = blockIdx.x;  // b*512 + t
  const int b = bt >> 9, t = bt & 511;
  __shared__ float rr[32], ri[32], amp[32], csc[8];
  __shared__ float gr[8][32], gi[8][32];
  __shared__ float twc[64], tws[64];
  const int tid = threadIdx.x;  // 256
  if (tid < 64) {
    float th = (6.28318530717958647f / 64.f) * (float)tid;  // + sign (inverse)
    __sincosf(th, &tws[tid], &twc[tid]);
  }
  if (tid < 32) {
    float r = rf_re[bt * 32 + tid], im = rf_im[bt * 32 + tid];
    rr[tid] = r; ri[tid] = im; amp[tid] = sqrtf(r * r + im * im);
  }
  if (tid < 8) csc[tid] = fabsf(c_h[tid]) * 0.25f;  // 1/sqrt(dh)=1/4
  __syncthreads();
  {
    int h = tid >> 5, m = tid & 31;
    float scale = csc[h] * amp[m];
    float mx = -1e30f;
    for (int n2 = 0; n2 < 32; ++n2) mx = fmaxf(mx, scale * amp[n2]);
    float sum = 0.f, ar = 0.f, ai = 0.f;
    for (int n2 = 0; n2 < 32; ++n2) {
      float w = __expf(scale * amp[n2] - mx);
      sum += w; ar += w * rr[n2]; ai += w * ri[n2];
    }
    gr[h][m] = ar / sum;
    gi[h][m] = ai / sum;
  }
  __syncthreads();
  for (int nn = tid >> 3; nn < 64; nn += 32) {
    int h = tid & 7;
    float acc = gr[h][0];
    for (int m2 = 1; m2 < 32; ++m2) {
      int idx = (m2 * nn) & 63;
      acc += 2.f * (gr[h][m2] * twc[idx] - gi[h][m2] * tws[idx]);
    }
    G[(((b * 64 + nn) * 8 + h) * 512) + t] = acc * (1.f / 64.f);
  }
}

// ---------------------------------------------------------------------------
// Kernel 5: twiddle matrices + bias vectors for the DFT GEMMs
// ---------------------------------------------------------------------------
__global__ void init_tables_kernel(const float* __restrict__ dr_b,
                                   float* __restrict__ Wf, float* __restrict__ Wi,
                                   float* __restrict__ bias0,
                                   float* __restrict__ biasdr) {
  const int gid = blockIdx.x * 256 + threadIdx.x;
  const int stride = gridDim.x * 256;
  for (int i = gid; i < 512 * 128; i += stride) {
    int t = i >> 7, col = i & 127, tau = col >> 1;
    float th = -(6.28318530717958647f / 512.f) * (float)((t * tau) & 511);
    float s, c;
    __sincosf(th, &s, &c);
    Wf[i] = (col & 1) ? s : c;
  }
  for (int i = gid; i < 128 * 512; i += stride) {
    int row = i >> 9, t = i & 511;
    int tau = row & 63;
    float th = (6.28318530717958647f / 512.f) * (float)((t * tau) & 511);
    float s, c;
    __sincosf(th, &s, &c);
    float v;
    if (row < 64) v = (tau == 0 ? 1.f : 2.f * c) * (1.f / 512.f);
    else          v = (tau == 0 ? 0.f : -2.f * s) * (1.f / 512.f);
    Wi[i] = v;
  }
  if (gid < 128) bias0[gid] = 0.f;
  if (gid < 512) biasdr[gid] = dr_b[0];
}

// ---------------------------------------------------------------------------
// Kernel 6: TEM complex attention (rank-8 collapsed) per (b,n) slot.
// ---------------------------------------------------------------------------
__global__ void tem_attn_kernel(const float* __restrict__ GF_ri,
                                const float* __restrict__ Pq,
                                const float* __restrict__ Pk,
                                const float* __restrict__ Dv,
                                float* __restrict__ z_ri) {
  extern __shared__ unsigned char smem_raw[];
  float* gfr = (float*)smem_raw;   // 512
  float* gfi = gfr + 512;          // 512
  float* pq  = gfi + 512;          // 1024
  float* pk  = pq + 1024;          // 1024
  float* dv  = pk + 1024;          // 64
  float* vzr = dv + 64;            // 512
  float* vzi = vzr + 512;          // 512
  float* zr  = vzi + 512;          // 64
  float* zi  = zr + 64;            // 64
  bf16* krs  = (bf16*)(zi + 64);   // 8192 bf16
  bf16* kis  = krs + 8192;         // 8192 bf16

  const int slot = blockIdx.x;
  const int tid = threadIdx.x;  // 256
  for (int i = tid; i < 512; i += 256) {
    int h = i >> 6, m = i & 63;  // m == tau
    gfr[m * 8 + h] = GF_ri[(slot * 8 + h) * 128 + 2 * m];
    gfi[m * 8 + h] = GF_ri[(slot * 8 + h) * 128 + 2 * m + 1];
  }
  for (int i = tid; i < 1024; i += 256) { pq[i] = Pq[i]; pk[i] = Pk[i]; }
  if (tid < 64) { dv[tid] = Dv[tid]; zr[tid] = 0.f; zi[tid] = 0.f; }
  __syncthreads();
  for (int i = tid; i < 8192; i += 256) {  // k planes: 64 x 128 (bf16)
    int n = i >> 7, d = i & 127;
    float ar = 0.f, ai = 0.f;
#pragma unroll
    for (int h1 = 0; h1 < 8; ++h1) {
      float w = pk[h1 * 128 + d];
      ar += gfr[n * 8 + h1] * w;
      ai += gfi[n * 8 + h1] * w;
    }
    krs[i] = f2bf(ar);
    kis[i] = f2bf(ai);
  }
  for (int i = tid; i < 512; i += 256) {  // vz: 64 x 8
    int n = i >> 3, h = i & 7;
    float ar = 0.f, ai = 0.f;
#pragma unroll
    for (int h1 = 0; h1 < 8; ++h1) {
      float w = dv[h * 8 + h1];
      ar += gfr[n * 8 + h1] * w;
      ai += gfi[n * 8 + h1] * w;
    }
    vzr[i] = ar; vzi[i] = ai;
  }
  __syncthreads();
  for (int iter = 0; iter < 2; ++iter) {
    int pair = iter * 256 + tid;  // 512 = 8 heads x 64 rows
    int h = pair >> 6, m = pair & 63;
    float qrd[16], qid[16];
#pragma unroll
    for (int d = 0; d < 16; ++d) {
      float ar = 0.f, ai = 0.f;
#pragma unroll
      for (int h1 = 0; h1 < 8; ++h1) {
        float w = pq[h1 * 128 + h * 16 + d];
        ar += gfr[m * 8 + h1] * w;
        ai += gfi[m * 8 + h1] * w;
      }
      qrd[d] = ar; qid[d] = ai;
    }
    float a[64];
    float mx = -1e30f;
    for (int n2 = 0; n2 < 64; ++n2) {
      float sr = 0.f, si = 0.f;
#pragma unroll
      for (int d = 0; d < 16; ++d) {
        float kv = (float)krs[n2 * 128 + h * 16 + d];
        float kw = (float)kis[n2 * 128 + h * 16 + d];
        sr += qrd[d] * kv + qid[d] * kw;
        si += qid[d] * kv - qrd[d] * kw;
      }
      float ab = sqrtf(sr * sr + si * si) * 0.25f;
      a[n2] = ab;
      mx = fmaxf(mx, ab);
    }
    float sum = 0.f, accr = 0.f, acci = 0.f;
    for (int n2 = 0; n2 < 64; ++n2) {
      float w = __expf(a[n2] - mx);
      sum += w;
      accr += w * vzr[n2 * 8 + h];
      acci += w * vzi[n2 * 8 + h];
    }
    float inv = 1.f / sum;
    atomicAdd(&zr[m], accr * inv);  // ds_add_f32
    atomicAdd(&zi[m], acci * inv);
  }
  __syncthreads();
  if (tid < 64) {
    z_ri[slot * 128 + tid] = zr[tid];
    z_ri[slot * 128 + 64 + tid] = zi[tid];
  }
}

// ---------------------------------------------------------------------------
// Kernel 7: generic WMMA GEMM  C = act(A(MxK) @ B(KxN) + bias)
//   TDM stages raw f32 tiles into LDS asynchronously (wave 0 programs the
//   descriptors); threads convert+swizzle into per-lane fragment order;
//   each wave fetches operands with 2x ds_load_b128 per fragment.
// ---------------------------------------------------------------------------
template <typename T>
__global__ void gemm_bias_act_kernel(const float* __restrict__ A,
                                     const float* __restrict__ B,
                                     const float* __restrict__ bias,
                                     float* __restrict__ C,
                                     int M, int N, int K, int act) {
  __shared__ __align__(16) unsigned As_pack[4 * 2 * 32 * 4];  // 4KB
  __shared__ __align__(16) unsigned Bs_pack[4 * 2 * 32 * 4];  // 4KB
#if HAVE_TDM
  __shared__ __align__(16) float Araw[64 * 32];               // 8KB staging
  __shared__ __align__(16) float Braw[32 * 64];               // 8KB staging
#endif
  const int tid = threadIdx.x;
  const int lane = tid & 31;
  const int wave = tid >> 5;            // 0..7
  const int bm = blockIdx.x * 64;
  const int bn = blockIdx.y * 64;
  const int tm = wave & 3;              // M sub-tile
  const int tn0 = (wave >> 2) * 2;      // N sub-tiles tn0, tn0+1

  v8f acc0 = {};
  v8f acc1 = {};
  const v4u* As4 = (const v4u*)As_pack;
  const v4u* Bs4 = (const v4u*)Bs_pack;

  for (int k0 = 0; k0 < K; k0 += 32) {
#if HAVE_TDM
    if (wave == 0) {
      // async DMA: A tile 32x64 rows, B tile 64x32 rows (zero-fill OOB)
      tdm_load_2d(&A[(size_t)bm * K + k0], lds_offset_of(Araw),
                  /*tile_w=*/32, /*tile_h=*/64,
                  /*tens_w=*/(unsigned)(K - k0), /*tens_h=*/(unsigned)(M - bm),
                  /*row_stride=*/(unsigned)K);
      tdm_load_2d(&B[(size_t)k0 * N + bn], lds_offset_of(Braw),
                  /*tile_w=*/64, /*tile_h=*/32,
                  /*tens_w=*/(unsigned)(N - bn), /*tens_h=*/(unsigned)(K - k0),
                  /*row_stride=*/(unsigned)N);
      __builtin_amdgcn_s_wait_tensorcnt(0);
    }
    __syncthreads();  // raw tiles visible to all waves
#else
    if (k0 + 32 < K) {  // global_prefetch_b8 next tiles
      __builtin_prefetch(&A[(size_t)(bm + (tid & 63)) * K + k0 + 32], 0, 0);
      __builtin_prefetch(&B[(size_t)(k0 + 32 + (tid & 31)) * N + bn], 0, 0);
    }
#endif
    // stage A tile 64x32: element (r,k) -> lane m+16g, frag elem 8c+p
    //   (g = (k>>3)&1, c = k>>4, p = k&7)  [CDNA5 16-bit A layout]
    for (int j = tid; j < 1024; j += 256) {
      int r = j >> 4;
      int k = (j & 15) << 1;
#if HAVE_TDM
      float a0 = Araw[r * 32 + k];
      float a1 = Araw[r * 32 + k + 1];
#else
      int gr = bm + r, gc = k0 + k;
      float a0 = (gr < M && gc < K) ? A[(size_t)gr * K + gc] : 0.f;
      float a1 = (gr < M && (gc + 1) < K) ? A[(size_t)gr * K + gc + 1] : 0.f;
#endif
      int tmw = r >> 4, m = r & 15;
      int g = (k >> 3) & 1, c = k >> 4, p = k & 7;
      int cell = (tmw * 2 + c) * 32 + m + (g << 4);
      As_pack[cell * 4 + (p >> 1)] = pack2<T>(a0, a1);
    }
    // stage B tile 32x64: element (k,n) -> lane n+16g, frag elem 8c+p
    //   (g = (k>>3)&1, c = k>>4, p = k&7)  [CDNA5 16-bit B layout]
    for (int j = tid; j < 1024; j += 256) {
      int n = j >> 4;
      int k = (j & 15) << 1;
#if HAVE_TDM
      float b0 = Braw[k * 64 + n];
      float b1 = Braw[(k + 1) * 64 + n];
#else
      int gr = k0 + k, gc = bn + n;
      float b0 = (gr < K && gc < N) ? B[(size_t)gr * N + gc] : 0.f;
      float b1 = ((gr + 1) < K && gc < N) ? B[(size_t)(gr + 1) * N + gc] : 0.f;
#endif
      int tnw = n >> 4, nl = n & 15;
      int g = (k >> 3) & 1, c = k >> 4, p = k & 7;
      int cell = (tnw * 2 + c) * 32 + nl + (g << 4);
      Bs_pack[cell * 4 + (p >> 1)] = pack2<T>(b0, b1);
    }
    __syncthreads();
    Frag<T> fa, fb0, fb1;
    fa.u[0]  = As4[(tm * 2 + 0) * 32 + lane];
    fa.u[1]  = As4[(tm * 2 + 1) * 32 + lane];
    fb0.u[0] = Bs4[(tn0 * 2 + 0) * 32 + lane];
    fb0.u[1] = Bs4[(tn0 * 2 + 1) * 32 + lane];
    fb1.u[0] = Bs4[((tn0 + 1) * 2 + 0) * 32 + lane];
    fb1.u[1] = Bs4[((tn0 + 1) * 2 + 1) * 32 + lane];
    acc0 = wmma_any(fa.v, fb0.v, acc0);
    acc1 = wmma_any(fa.v, fb1.v, acc1);
    __syncthreads();  // pack + staging buffers free for next iteration
  }
  const int n_local = lane & 15;
  const int m_hi = (lane >> 4) * 8;
#pragma unroll
  for (int r = 0; r < 8; ++r) {
    int gm = bm + tm * 16 + r + m_hi;
    int gn0 = bn + tn0 * 16 + n_local;
    int gn1 = bn + (tn0 + 1) * 16 + n_local;
    if (gm < M && gn0 < N) {
      float v = acc0[r] + bias[gn0];
      if (act == 1) v = v > 0.f ? v : 0.f;
      C[(size_t)gm * N + gn0] = v;
    }
    if (gm < M && gn1 < N) {
      float v = acc1[r] + bias[gn1];
      if (act == 1) v = v > 0.f ? v : 0.f;
      C[(size_t)gm * N + gn1] = v;
    }
  }
}

// ---------------------------------------------------------------------------
// Kernel 8: fuse + transpose: out[b,p,n] = Wf0*xo[b,n,p] + Wf1*trend[b,n,p]
// ---------------------------------------------------------------------------
__global__ void fuse_kernel(const float* __restrict__ xo,
                            const float* __restrict__ trend_bn,
                            const float* __restrict__ W_fuse,
                            float* __restrict__ out) {
  int i = blockIdx.x * 256 + threadIdx.x;
  if (i >= 8 * 96 * 64) return;
  int n = i & 63;
  int p = (i >> 6) % 96;
  int b = i / (96 * 64);
  int src = (b * 64 + n) * 96 + p;
  out[i] = W_fuse[0] * xo[src] + W_fuse[1] * trend_bn[src];
}

// ---------------------------------------------------------------------------
extern "C" void kernel_launch(void* const* d_in, const int* in_sizes, int n_in,
                              void* d_out, int out_size, void* d_ws, size_t ws_size,
                              hipStream_t stream) {
  const float* x      = (const float*)d_in[0];
  const float* emb    = (const float*)d_in[1];
  const float* dec_w  = (const float*)d_in[2];
  const float* dec_b  = (const float*)d_in[3];
  const float* mlp_w1 = (const float*)d_in[4];
  const float* mlp_b1 = (const float*)d_in[5];
  const float* mlp_w2 = (const float*)d_in[6];
  const float* mlp_b2 = (const float*)d_in[7];
  const float* mlp_w3 = (const float*)d_in[8];
  const float* mlp_b3 = (const float*)d_in[9];
  const float* cWq = (const float*)d_in[10];
  const float* cWk = (const float*)d_in[11];
  const float* cWv = (const float*)d_in[12];
  const float* cWo = (const float*)d_in[13];
  const float* tWq = (const float*)d_in[14];
  const float* tWk = (const float*)d_in[15];
  const float* tWv = (const float*)d_in[16];
  const float* tWo = (const float*)d_in[17];
  const float* dr_w = (const float*)d_in[18];
  const float* dr_b = (const float*)d_in[19];
  const float* out_w = (const float*)d_in[20];
  const float* out_b = (const float*)d_in[21];
  const float* W_fuse = (const float*)d_in[22];
  float* out = (float*)d_out;
  (void)in_sizes; (void)n_in; (void)out_size; (void)ws_size;

  float* ws = (float*)d_ws;
  size_t off = 0;
  auto alloc = [&](size_t nfl) { float* p = ws + off; off += nfl; return p; };
  float* res      = alloc(8 * 512 * 64);   // (B,T,N)
  float* trendT   = alloc(512 * 512);      // (B*N, T)
  float* H1       = alloc(512 * 256);
  float* H2       = alloc(512 * 256);
  float* trend_bn = alloc(512 * 96);
  float* rf_re    = alloc(4096 * 32);
  float* rf_im    = alloc(4096 * 32);
  float* c8       = alloc(8);
  float* Pq       = alloc(8 * 128);
  float* Pk       = alloc(8 * 128);
  float* Dv       = alloc(64);
  float* G        = alloc(4096 * 512);     // rows (b,n,h), cols t
  float* GF_ri    = alloc(4096 * 128);     // rows (b,n,h), cols 2*tau+{re,im}
  float* z_ri     = alloc(512 * 128);      // rows (b,n), cols [re0..63|im0..63]
  float* xr       = alloc(512 * 512);      // (B*N, T)
  float* xo       = alloc(512 * 96);
  float* Wf       = alloc(512 * 128);      // forward twiddles
  float* Wi       = alloc(128 * 512);      // inverse twiddles
  float* bias0    = alloc(128);
  float* biasdr   = alloc(512);

  init_tables_kernel<<<128, 256, 0, stream>>>(dr_b, Wf, Wi, bias0, biasdr);
  decomp_kernel<<<512, 128, 0, stream>>>(x, dec_w, dec_b, res, trendT);
  rfft64_kernel<<<512, 256, 0, stream>>>(res, rf_re, rf_im);
  prep_kernel<<<1, 128, 0, stream>>>(emb, cWq, cWk, cWv, cWo, tWq, tWk, tWv, tWo,
                                     dr_w, c8, Pq, Pk, Dv);
  cem_attn_kernel<<<4096, 256, 0, stream>>>(rf_re, rf_im, c8, G);

  // 512-pt rDFT over t as WMMA-f16 GEMM: GF = G @ Wf
  gemm_bias_act_kernel<_Float16><<<dim3(64, 2), 256, 0, stream>>>(
      G, Wf, bias0, GF_ri, 4096, 128, 512, 0);

  size_t smemT = (4288 * sizeof(float)) + (2 * 8192 * sizeof(bf16));  // 49920 B
  tem_attn_kernel<<<512, 256, smemT, stream>>>(GF_ri, Pq, Pk, Dv, z_ri);

  // 512-pt irDFT + dr reduction bias as WMMA-f16 GEMM: xr = Z @ Wi + dr_b
  gemm_bias_act_kernel<_Float16><<<dim3(8, 8), 256, 0, stream>>>(
      z_ri, Wi, biasdr, xr, 512, 512, 128, 0);

  // trend MLP stack + output head (WMMA-bf16)
  gemm_bias_act_kernel<bf16><<<dim3(8, 4), 256, 0, stream>>>(
      trendT, mlp_w1, mlp_b1, H1, 512, 256, 512, 1);
  gemm_bias_act_kernel<bf16><<<dim3(8, 4), 256, 0, stream>>>(
      H1, mlp_w2, mlp_b2, H2, 512, 256, 256, 1);
  gemm_bias_act_kernel<bf16><<<dim3(8, 2), 256, 0, stream>>>(
      H2, mlp_w3, mlp_b3, trend_bn, 512, 96, 256, 0);
  gemm_bias_act_kernel<bf16><<<dim3(8, 2), 256, 0, stream>>>(
      xr, out_w, out_b, xo, 512, 96, 512, 0);
  fuse_kernel<<<192, 256, 0, stream>>>(xo, trend_bn, W_fuse, out);
}
